// MLPAC_4_team_49263274885829
// MI455X (gfx1250) — compile-verified, matrix operand order checked by reading, not run on hardware
//
#include <hip/hip_runtime.h>
#include <hip/hip_bf16.h>
#include <math.h>
#include <stdint.h>

typedef __attribute__((ext_vector_type(16))) _Float16 v16h;
typedef __attribute__((ext_vector_type(8)))  _Float16 v8h;
typedef __attribute__((ext_vector_type(8)))  float    v8f;
typedef __attribute__((ext_vector_type(4)))  unsigned int u32x4;
typedef __attribute__((ext_vector_type(8)))  int      i32x8;
typedef __attribute__((ext_vector_type(4)))  int      i32x4;

#define P_NUM     3
#define OBS_N     48
#define ROWS      64
#define NTHREADS  256
#define NEG_SLOPE 0.01f
#define ACT_LIMIT 1.0f

// LDS strides in halves (16B-aligned rows; row stride != 0 mod 64 banks)
#define H1F_STRIDE 392   // 384 feature cols + 8 pad
#define ENC_STRIDE 584   // 576 + 8
#define H_STRIDE   264   // 256 + 8

#define OFF_A   0
#define SZ_A    (ROWS * H1F_STRIDE * 2)          // 50176 : h1feat (ph1-2) / bufH2 (ph4-5)
#define OFF_ENC (OFF_A + SZ_A)
#define SZ_ENC  (ROWS * ENC_STRIDE * 2)          // 74752
#define OFF_H1  (OFF_ENC + SZ_ENC)
#define SZ_H1   (ROWS * H_STRIDE * 2)            // 33792
#define OFF_OBS (OFF_H1 + SZ_H1)                 // 158720
#define SZ_OBS  (ROWS * OBS_N * 4)               // 12288
#define SMEM_BYTES (OFF_OBS + SZ_OBS)            // 171008 bytes < 320KB

// Workspace layout (f16, transposed weights Wt[N][K])
#define NE_WTE (3 * 576 * 96)                    // [p][enc col c][3 encoders][32 k]
#define N_WT1  (3 * 256 * 576)                   // [p][n][k]
#define N_WT2  (3 * 256 * 256)                   // [p][n][k]
#define WS_NEEDED ((size_t)(NE_WTE + N_WT1 + N_WT2) * 2)

#if defined(__has_builtin)
#if __has_builtin(__builtin_amdgcn_tensor_load_to_lds)
#define HAVE_TDM 1
#endif
#endif
#ifndef HAVE_TDM
#define HAVE_TDM 0
#endif

__device__ __forceinline__ float lrelu(float x) {
    return x > 0.0f ? x : NEG_SLOPE * x;
}

__device__ __forceinline__ v8f splat8(float x) {
    v8f v;
#pragma unroll
    for (int i = 0; i < 8; ++i) v[i] = x;
    return v;
}

// Fragment load from f16 row-major [rows][stride_h] (LDS or global):
// lane: row = lane&15, kbase = (lane<16)?0:8 ; halves = K {kb..kb+7, kb+16..kb+23}
__device__ __forceinline__ v16h load_frag16(const _Float16* base, int stride_h, int lane) {
    const int m  = lane & 15;
    const int kb = (lane < 16) ? 0 : 8;
    const _Float16* p0 = base + m * stride_h + kb;
    union { v16h v; v8h h[2]; } f;
    f.h[0] = *(const v8h*)(p0);        // 16B chunk (b128)
    f.h[1] = *(const v8h*)(p0 + 16);
    return f.v;
}

// Fallback: B-fragment gathered from global row-major f32 W[32][ldn]
__device__ __forceinline__ v16h load_bfrag_f32(const float* W, int ldn, int ncol, int lane) {
    const int kb = (lane < 16) ? 0 : 8;
    union { v16h v; _Float16 e[16]; } f;
#pragma unroll
    for (int h = 0; h < 8; ++h) {
        f.e[h]     = (_Float16)W[(kb + h)      * ldn + ncol];
        f.e[h + 8] = (_Float16)W[(kb + h + 16) * ldn + ncol];
    }
    return f.v;
}

__device__ __forceinline__ v8f wmma_f32_f16(v16h a, v16h b, v8f c) {
    return __builtin_amdgcn_wmma_f32_16x16x32_f16(false, a, false, b, (short)0, c, false, false);
}

// ---------------------------------------------------------------------------
// Prep kernel: one-shot convert+transpose of all matmul weights into ws (f16)
// ---------------------------------------------------------------------------
__global__ void prep_weights_kernel(
    const float* __restrict__ prop_W2, const float* __restrict__ ext_W2,
    const float* __restrict__ opp_W2,  const float* __restrict__ pi_W1,
    const float* __restrict__ pi_W2,
    _Float16* __restrict__ wtE, _Float16* __restrict__ wt1, _Float16* __restrict__ wt2)
{
    const int idx = blockIdx.x * NTHREADS + threadIdx.x;
    if (idx < NE_WTE) {
        const int p = idx / (576 * 96);
        const int rem = idx % (576 * 96);
        const int c = rem / 96;
        const int e = (rem % 96) >> 5;
        const int k = rem & 31;
        float v;
        if (e == 0)      v = prop_W2[((p * 9 + (c >> 6)) * 32 + k) * 64 + (c & 63)];
        else if (e == 1) v = ext_W2[((p * 2 + (c / 288)) * 32 + k) * 288 + (c % 288)];
        else             v = opp_W2[(p * 32 + k) * 576 + c];
        wtE[idx] = (_Float16)v;
    } else if (idx < NE_WTE + N_WT1) {
        const int t = idx - NE_WTE;
        const int p = t / (576 * 256);
        const int r = t % (576 * 256);
        const int k = r >> 8, n = r & 255;              // coalesced read over n
        wt1[((size_t)p * 256 + n) * 576 + k] = (_Float16)pi_W1[((size_t)p * 576 + k) * 256 + n];
    } else if (idx < NE_WTE + N_WT1 + N_WT2) {
        const int t = idx - NE_WTE - N_WT1;
        const int p = t / 65536;
        const int r = t & 65535;
        const int k = r >> 8, n = r & 255;
        wt2[((size_t)p * 256 + n) * 256 + k] = (_Float16)pi_W2[((size_t)p * 256 + k) * 256 + n];
    }
}

// ---------------------------------------------------------------------------
// Fused MLP actor kernel
// ---------------------------------------------------------------------------
template<bool USE_WS>
__global__ void mlpac_fused_kernel(
    const float* __restrict__ obs,
    const float* __restrict__ prop_W1, const float* __restrict__ prop_b1,
    const float* __restrict__ prop_W2, const float* __restrict__ prop_b2,
    const float* __restrict__ ext_W1,  const float* __restrict__ ext_b1,
    const float* __restrict__ ext_W2,  const float* __restrict__ ext_b2,
    const float* __restrict__ opp_W1,  const float* __restrict__ opp_b1,
    const float* __restrict__ opp_W2,  const float* __restrict__ opp_b2,
    const float* __restrict__ pi_W1,   const float* __restrict__ pi_b1,
    const float* __restrict__ pi_W2,   const float* __restrict__ pi_b2,
    const float* __restrict__ pi_W3,   const float* __restrict__ pi_b3,
    const _Float16* __restrict__ wtE,  const _Float16* __restrict__ wt1,
    const _Float16* __restrict__ wt2,
    float* __restrict__ out)
{
    extern __shared__ __align__(16) char smem[];
    _Float16* h1feat = (_Float16*)(smem + OFF_A);    // phases 1-2
    _Float16* bufH2  = (_Float16*)(smem + OFF_A);    // phases 4-5 (reuse)
    _Float16* bufEnc = (_Float16*)(smem + OFF_ENC);  // phases 2-3
    _Float16* bufH1  = (_Float16*)(smem + OFF_H1);   // phases 3-4
    float*    ldsobs = (float*)   (smem + OFF_OBS);  // obs tile (TDM dest)

    const int tid  = threadIdx.x;
    const int lane = tid & 31;
    const int wv   = tid >> 5;        // 0..7
    const int l16  = lane & 15;
    const int hi   = lane >> 4;       // 0|1 -> +8 row offset for C/D layout
    const int p    = blockIdx.y;      // player
    const long rowbase = (long)blockIdx.x * ROWS;

    // ---------------- Phase 0: DMA obs tile (64 rows x 48 f32) into LDS --------
    const float* gobs = obs + ((size_t)rowbase * P_NUM + p) * OBS_N;
#if HAVE_TDM
    if (wv == 0) {
        // Tensor DMA descriptor: 2-D, data_size=4B, tile 48x64, row stride 144 elems
        const unsigned long long ga = (unsigned long long)(uintptr_t)gobs;
        u32x4 g0 = { 1u,                                   // count=1 (valid D#)
                     (unsigned)OFF_OBS,                    // lds_addr (bytes)
                     (unsigned)(ga & 0xFFFFFFFFull),       // global_addr[31:0]
                     (unsigned)((ga >> 32) & 0x01FFFFFFull) | 0x80000000u }; // addr[56:32] | type=2
        i32x8 g1 = { 0x00020000,        // wg_mask=0, data_size=2 (4B)
                     (48 << 16),        // tensor_dim0 = 48 (low bits in [63:48])
                     0,                 // tensor_dim0 hi / tensor_dim1 lo
                     (48 << 16) | 1,    // tensor_dim1 = 65536, tile_dim0 = 48
                     64,                // tile_dim1 = 64, tile_dim2 = 0
                     144,               // tensor_dim0_stride = 144 elements
                     0, 0 };            // tensor_dim1_stride = 0 (2-D tile)
        i32x4 gz  = { 0, 0, 0, 0 };
        i32x8 gz8 = { 0, 0, 0, 0, 0, 0, 0, 0 };
        __builtin_amdgcn_tensor_load_to_lds(g0, g1, gz, gz, gz8, 0);
        __builtin_amdgcn_s_wait_tensorcnt(0);
    }
#else
    for (int idx = tid; idx < ROWS * OBS_N; idx += NTHREADS)
        ldsobs[idx] = gobs[(idx / OBS_N) * (P_NUM * OBS_N) + (idx % OBS_N)];
#endif
    __syncthreads();

    // ---------------- Phase 1: encoder first layers (tiny K: 2/6/18) -> VALU ----
    {
        const int r = tid & 63;                 // row within tile
        const int g = tid >> 6;                 // 0..3 -> units g, g+4, g+8
        const float* orow = ldsobs + r * OBS_N;
#pragma unroll
        for (int uu = 0; uu < 3; ++uu) {
            const int u = g + uu * 4;           // 0..11
            if (u < 9) {                        // prop unit u : K=2 -> 32
                const float* W  = prop_W1 + ((p * 9 + u) * 2) * 32;
                const float* bb = prop_b1 + (p * 9 + u) * 32;
                const float x0 = orow[u * 2 + 0], x1 = orow[u * 2 + 1];
#pragma unroll 8
                for (int o = 0; o < 32; ++o) {
                    float v = fmaf(x0, W[o], fmaf(x1, W[32 + o], bb[o]));
                    h1feat[r * H1F_STRIDE + u * 32 + o] = (_Float16)lrelu(v);
                }
            } else if (u < 11) {                // ext unit e : K=6 -> 32
                const int e = u - 9;
                const float* W  = ext_W1 + ((p * 2 + e) * 6) * 32;
                const float* bb = ext_b1 + (p * 2 + e) * 32;
                float x[6];
#pragma unroll
                for (int i = 0; i < 6; ++i) x[i] = orow[18 + e * 6 + i];
#pragma unroll 4
                for (int o = 0; o < 32; ++o) {
                    float v = bb[o];
#pragma unroll
                    for (int i = 0; i < 6; ++i) v = fmaf(x[i], W[i * 32 + o], v);
                    h1feat[r * H1F_STRIDE + 288 + e * 32 + o] = (_Float16)lrelu(v);
                }
            } else {                            // opp : K=18 -> 32
                const float* W  = opp_W1 + p * 18 * 32;
                const float* bb = opp_b1 + p * 32;
                float x[18];
#pragma unroll
                for (int i = 0; i < 18; ++i) x[i] = orow[30 + i];
#pragma unroll 2
                for (int o = 0; o < 32; ++o) {
                    float v = bb[o];
#pragma unroll
                    for (int i = 0; i < 18; ++i) v = fmaf(x[i], W[i * 32 + o], v);
                    h1feat[r * H1F_STRIDE + 352 + o] = (_Float16)lrelu(v);
                }
            }
        }
    }
    __syncthreads();

    // ---------------- Phase 2: encoder second layers (3x K=32 WMMA per tile) ----
    // enc[64][576] = lrelu(propL2) + lrelu(extL2) + lrelu(oppL2)
    for (int j = wv; j < 36; j += 8) {          // 36 N-tiles of 16
        const int pu = j >> 2;                  // prop unit, 4 tiles each
        const int pn = (j & 3) * 16 + l16;      // col within prop unit (0..63)
        const int eu = j / 18;                  // ext unit, 18 tiles each
        const int en = (j - eu * 18) * 16 + l16;// col within ext unit (0..287)
        const int on = j * 16 + l16;            // opp col (0..575)

        v16h bp, be, bo;
        if constexpr (USE_WS) {
            const _Float16* eb = wtE + ((size_t)p * 576 + j * 16) * 96;
            bp = load_frag16(eb +  0, 96, lane);
            be = load_frag16(eb + 32, 96, lane);
            bo = load_frag16(eb + 64, 96, lane);
        } else {
            bp = load_bfrag_f32(prop_W2 + (p * 9 + pu) * 32 * 64, 64, pn, lane);
            be = load_bfrag_f32(ext_W2 + (p * 2 + eu) * 32 * 288, 288, en, lane);
            bo = load_bfrag_f32(opp_W2 + p * 32 * 576, 576, on, lane);
        }
        const float biasP = prop_b2[(p * 9 + pu) * 64 + pn];
        const float biasE = ext_b2[(p * 2 + eu) * 288 + en];
        const float biasO = opp_b2[p * 576 + on];

#pragma unroll
        for (int m = 0; m < 4; ++m) {
            v16h ap = load_frag16(h1feat + (m * 16) * H1F_STRIDE + pu * 32, H1F_STRIDE, lane);
            v16h ae = load_frag16(h1feat + (m * 16) * H1F_STRIDE + 288 + eu * 32, H1F_STRIDE, lane);
            v16h ao = load_frag16(h1feat + (m * 16) * H1F_STRIDE + 352, H1F_STRIDE, lane);
            v8f cp = wmma_f32_f16(ap, bp, splat8(biasP));
            v8f ce = wmma_f32_f16(ae, be, splat8(biasE));
            v8f co = wmma_f32_f16(ao, bo, splat8(biasO));
#pragma unroll
            for (int r = 0; r < 8; ++r) {
                float v = lrelu(cp[r]) + lrelu(ce[r]) + lrelu(co[r]);
                const int row = m * 16 + r + hi * 8;
                bufEnc[row * ENC_STRIDE + j * 16 + l16] = (_Float16)v;
            }
        }
    }
    __syncthreads();

    // ---------------- Phase 3: pi1 = lrelu(enc @ W1 + b1)  [576 -> 256] --------
    // Each wave owns all 64 rows x 32 cols: every B fragment fetched exactly once.
    const int ncol0 = wv * 32;
    {
        if constexpr (USE_WS)
            __builtin_prefetch(wt1 + ((size_t)p * 256 + ncol0) * 576, 0, 1);

        v8f acc[4][2];
#pragma unroll
        for (int jt = 0; jt < 2; ++jt) {
            const float bv = pi_b1[p * 256 + ncol0 + jt * 16 + l16];
#pragma unroll
            for (int m = 0; m < 4; ++m) acc[m][jt] = splat8(bv);
        }

        for (int ks = 0; ks < 18; ++ks) {       // K = 576, panels of 32
            v16h b0, b1;
            if constexpr (USE_WS) {
                const _Float16* wb = wt1 + ((size_t)p * 256 + ncol0) * 576 + ks * 32;
                b0 = load_frag16(wb, 576, lane);
                b1 = load_frag16(wb + 16 * 576, 576, lane);
            } else {
                const float* Wp = pi_W1 + (size_t)p * 576 * 256 + (size_t)ks * 32 * 256;
                b0 = load_bfrag_f32(Wp, 256, ncol0 + l16, lane);
                b1 = load_bfrag_f32(Wp, 256, ncol0 + 16 + l16, lane);
            }
#pragma unroll
            for (int m = 0; m < 4; ++m) {
                v16h a = load_frag16(bufEnc + (m * 16) * ENC_STRIDE + ks * 32, ENC_STRIDE, lane);
                acc[m][0] = wmma_f32_f16(a, b0, acc[m][0]);
                acc[m][1] = wmma_f32_f16(a, b1, acc[m][1]);
            }
        }
#pragma unroll
        for (int m = 0; m < 4; ++m)
#pragma unroll
            for (int jt = 0; jt < 2; ++jt)
#pragma unroll
                for (int r = 0; r < 8; ++r) {
                    const int row = m * 16 + r + hi * 8;
                    const int col = ncol0 + jt * 16 + l16;
                    bufH1[row * H_STRIDE + col] = (_Float16)lrelu(acc[m][jt][r]);
                }
    }
    __syncthreads();

    // ---------------- Phase 4: pi2 = lrelu(h1 @ W2 + b2)  [256 -> 256] ---------
    {
        v8f acc[4][2];
#pragma unroll
        for (int jt = 0; jt < 2; ++jt) {
            const float bv = pi_b2[p * 256 + ncol0 + jt * 16 + l16];
#pragma unroll
            for (int m = 0; m < 4; ++m) acc[m][jt] = splat8(bv);
        }

        for (int ks = 0; ks < 8; ++ks) {        // K = 256
            v16h b0, b1;
            if constexpr (USE_WS) {
                const _Float16* wb = wt2 + ((size_t)p * 256 + ncol0) * 256 + ks * 32;
                b0 = load_frag16(wb, 256, lane);
                b1 = load_frag16(wb + 16 * 256, 256, lane);
            } else {
                const float* Wp = pi_W2 + (size_t)p * 256 * 256 + (size_t)ks * 32 * 256;
                b0 = load_bfrag_f32(Wp, 256, ncol0 + l16, lane);
                b1 = load_bfrag_f32(Wp, 256, ncol0 + 16 + l16, lane);
            }
#pragma unroll
            for (int m = 0; m < 4; ++m) {
                v16h a = load_frag16(bufH1 + (m * 16) * H_STRIDE + ks * 32, H_STRIDE, lane);
                acc[m][0] = wmma_f32_f16(a, b0, acc[m][0]);
                acc[m][1] = wmma_f32_f16(a, b1, acc[m][1]);
            }
        }
        __syncthreads();   // bufH2 aliases h1feat region; also all bufH1 reads done
#pragma unroll
        for (int m = 0; m < 4; ++m)
#pragma unroll
            for (int jt = 0; jt < 2; ++jt)
#pragma unroll
                for (int r = 0; r < 8; ++r) {
                    const int row = m * 16 + r + hi * 8;
                    const int col = ncol0 + jt * 16 + l16;
                    bufH2[row * H_STRIDE + col] = (_Float16)lrelu(acc[m][jt][r]);
                }
    }
    __syncthreads();

    // ---------------- Phase 5: pi3 = tanh(h2 @ W3 + b3)  [256 -> 3] ------------
    if (tid < ROWS * 3) {
        const int r = tid / 3, a = tid - r * 3;
        const float* W = pi_W3 + p * 256 * 3;
        float acc = pi_b3[p * 3 + a];
#pragma unroll 8
        for (int k = 0; k < 256; ++k)
            acc = fmaf((float)bufH2[r * H_STRIDE + k], W[k * 3 + a], acc);
        const long grow = rowbase + r;
        out[(grow * P_NUM + p) * 3 + a] = ACT_LIMIT * tanhf(acc);
    }
}

extern "C" void kernel_launch(void* const* d_in, const int* in_sizes, int n_in,
                              void* d_out, int out_size, void* d_ws, size_t ws_size,
                              hipStream_t stream) {
    (void)n_in; (void)out_size;
    const float* obs     = (const float*)d_in[0];
    const float* prop_W1 = (const float*)d_in[1];
    const float* prop_b1 = (const float*)d_in[2];
    const float* prop_W2 = (const float*)d_in[3];
    const float* prop_b2 = (const float*)d_in[4];
    const float* ext_W1  = (const float*)d_in[5];
    const float* ext_b1  = (const float*)d_in[6];
    const float* ext_W2  = (const float*)d_in[7];
    const float* ext_b2  = (const float*)d_in[8];
    const float* opp_W1  = (const float*)d_in[9];
    const float* opp_b1  = (const float*)d_in[10];
    const float* opp_W2  = (const float*)d_in[11];
    const float* opp_b2  = (const float*)d_in[12];
    const float* pi_W1   = (const float*)d_in[13];
    const float* pi_b1   = (const float*)d_in[14];
    const float* pi_W2   = (const float*)d_in[15];
    const float* pi_b2   = (const float*)d_in[16];
    const float* pi_W3   = (const float*)d_in[17];
    const float* pi_b3   = (const float*)d_in[18];
    float* out = (float*)d_out;

    const int Btot = in_sizes[0] / (P_NUM * OBS_N);   // 65536
    dim3 grid(Btot / ROWS, P_NUM);

    const bool use_ws = (d_ws != nullptr) && (ws_size >= WS_NEEDED);
    _Float16* wtE = (_Float16*)d_ws;
    _Float16* wt1 = wtE ? wtE + NE_WTE : nullptr;
    _Float16* wt2 = wt1 ? wt1 + N_WT1  : nullptr;

    if (use_ws) {
        const int total = NE_WTE + N_WT1 + N_WT2;
        prep_weights_kernel<<<(total + NTHREADS - 1) / NTHREADS, NTHREADS, 0, stream>>>(
            prop_W2, ext_W2, opp_W2, pi_W1, pi_W2, wtE, wt1, wt2);
        (void)hipFuncSetAttribute((const void*)mlpac_fused_kernel<true>,
                                  hipFuncAttributeMaxDynamicSharedMemorySize, SMEM_BYTES);
        mlpac_fused_kernel<true><<<grid, NTHREADS, SMEM_BYTES, stream>>>(
            obs, prop_W1, prop_b1, prop_W2, prop_b2,
            ext_W1, ext_b1, ext_W2, ext_b2,
            opp_W1, opp_b1, opp_W2, opp_b2,
            pi_W1, pi_b1, pi_W2, pi_b2, pi_W3, pi_b3,
            wtE, wt1, wt2, out);
    } else {
        (void)hipFuncSetAttribute((const void*)mlpac_fused_kernel<false>,
                                  hipFuncAttributeMaxDynamicSharedMemorySize, SMEM_BYTES);
        mlpac_fused_kernel<false><<<grid, NTHREADS, SMEM_BYTES, stream>>>(
            obs, prop_W1, prop_b1, prop_W2, prop_b2,
            ext_W1, ext_b1, ext_W2, ext_b2,
            opp_W1, opp_b1, opp_W2, opp_b2,
            pi_W1, pi_b1, pi_W2, pi_b2, pi_W3, pi_b3,
            nullptr, nullptr, nullptr, out);
    }
}
